// Loss_CE_22376779612593
// MI455X (gfx1250) — matrix operand order repaired and checked by reading, not compiled, not run on hardware
//
#include <hip/hip_runtime.h>
#include <hip/hip_bf16.h>

typedef __attribute__((ext_vector_type(16))) _Float16 v16h;
typedef __attribute__((ext_vector_type(8)))  float    v8f;

#define N_ROWS 8192
#define DIM    256
#define TILE   128
#define KSTEP  32
#define LDS_STRIDE 40              // halves; 80B per row, multiple of 16B
#define BUF_BYTES (TILE * LDS_STRIDE * 2)   // 10240 B per tile buffer
#define TEMP_INV (1.0f / 0.07f)    // also the fixed logsumexp shift M (max |s|)

// ---------------------------------------------------------------------------
// CDNA5 async global->LDS DMA (ASYNCcnt path). vdst = per-lane LDS byte
// address (low 32 bits of generic shared pointer), vaddr = 64-bit global.
// ---------------------------------------------------------------------------
__device__ __forceinline__ void async_ld16(unsigned lds_byte, const void* gaddr) {
    asm volatile("global_load_async_to_lds_b128 %0, %1, off"
                 :: "v"(lds_byte), "v"(gaddr)
                 : "memory");
}

// ---------------------------------------------------------------------------
// Kernel 1: per-row L2-normalize both matrices into f16, and compute the
// diagonal similarity s_ii = cos(ecg_i, text_i)/T. One block (256 thr) per row.
// ---------------------------------------------------------------------------
__global__ __launch_bounds__(256) void norm_diag_kernel(
    const float* __restrict__ ecg, const float* __restrict__ text,
    _Float16* __restrict__ eN, _Float16* __restrict__ tN,
    float* __restrict__ diag)
{
    const int i = blockIdx.x;
    const int t = threadIdx.x;           // DIM == 256 == blockDim.x
    const size_t base = (size_t)i * DIM + t;

    const float e = ecg[base];
    const float x = text[base];
    float ee = e * e, xx = x * x, ex = e * x;

    #pragma unroll
    for (int m = 1; m < 32; m <<= 1) {
        ee += __shfl_xor(ee, m);
        xx += __shfl_xor(xx, m);
        ex += __shfl_xor(ex, m);
    }
    __shared__ float s_ee[8], s_xx[8], s_ex[8];
    const int w = t >> 5, lane = t & 31;
    if (lane == 0) { s_ee[w] = ee; s_xx[w] = xx; s_ex[w] = ex; }
    __syncthreads();

    float tee = 0.f, txx = 0.f, tex = 0.f;
    #pragma unroll
    for (int j = 0; j < 8; ++j) { tee += s_ee[j]; txx += s_xx[j]; tex += s_ex[j]; }

    const float re = rsqrtf(tee);
    const float rx = rsqrtf(txx);
    eN[base] = (_Float16)(e * re);
    tN[base] = (_Float16)(x * rx);
    if (t == 0) diag[i] = tex * re * rx * TEMP_INV;   // s_ii
}

// ---------------------------------------------------------------------------
// Kernel 2: tiled WMMA GEMM of normalized f16 rows (A @ B^T), double-buffered
// async-DMA staging, fused with exp(s - M) (diag excluded) and row/column
// partial-sum accumulation. Block = 256 threads = 8 waves; tile = 128x128.
// ---------------------------------------------------------------------------
__global__ __launch_bounds__(256) void ntxent_gemm_kernel(
    const _Float16* __restrict__ eN, const _Float16* __restrict__ tN,
    float* __restrict__ rowAcc, float* __restrict__ colAcc)
{
    __shared__ __align__(16) _Float16 As[2][TILE * LDS_STRIDE];
    __shared__ __align__(16) _Float16 Bs[2][TILE * LDS_STRIDE];
    __shared__ float colS[TILE];

    const int tid  = threadIdx.x;
    const int lane = tid & 31;
    const int w    = tid >> 5;
    const int tileM = blockIdx.y * TILE;
    const int tileN = blockIdx.x * TILE;

    if (tid < TILE) colS[tid] = 0.f;

    // LDS byte addresses for the DMA engine (low 32 bits of generic pointer).
    const unsigned ldsA = (unsigned)(uintptr_t)&As[0][0];
    const unsigned ldsB = (unsigned)(uintptr_t)&Bs[0][0];

    // Per-thread staging assignments: two 16B chunks per matrix per buffer.
    // a = tid + r*256 in 0..511 -> row = a>>2, chunk = (a&3)*8 halves.
    const int row0 = tid >> 2,          ch0 = (tid & 3) * 8;
    const int row1 = (tid + 256) >> 2,  ch1 = ch0;           // (a&3) identical
    const unsigned lo0 = (unsigned)((row0 * LDS_STRIDE + ch0) * 2);
    const unsigned lo1 = (unsigned)((row1 * LDS_STRIDE + ch1) * 2);
    const _Float16* gA0 = eN + (size_t)(tileM + row0) * DIM + ch0;
    const _Float16* gA1 = eN + (size_t)(tileM + row1) * DIM + ch1;
    const _Float16* gB0 = tN + (size_t)(tileN + row0) * DIM + ch0;
    const _Float16* gB1 = tN + (size_t)(tileN + row1) * DIM + ch1;

    // Kick off buffer 0 (k0 = 0): 4 async DMA instructions per wave.
    async_ld16(ldsA + lo0, gA0);
    async_ld16(ldsA + lo1, gA1);
    async_ld16(ldsB + lo0, gB0);
    async_ld16(ldsB + lo1, gB1);

    const v8f zero8 = {0.f, 0.f, 0.f, 0.f, 0.f, 0.f, 0.f, 0.f};
    v8f acc[8];
    #pragma unroll
    for (int n = 0; n < 8; ++n) acc[n] = zero8;

    // Fragment addressing per CDNA5 16-bit WMMA VGPR layouts.
    const int ar    = 16 * w + (lane & 15);          // A row within tile
    const int abase = (lane < 16) ? 0 : 8;           // A: K halves {0..7,16..23}/{8..15,24..31}
    const int bbase = (lane < 16) ? 0 : 8;           // B: K halves {0..7,16..23}/{8..15,24..31}

    #pragma unroll
    for (int kk = 0; kk < DIM / KSTEP; ++kk) {
        const int cur = kk & 1;
        __syncthreads();   // all waves done reading buf[cur^1] (iter kk-1)

        if (kk + 1 < DIM / KSTEP) {
            const unsigned boff = (unsigned)((cur ^ 1) * BUF_BYTES);
            const int knext = (kk + 1) * KSTEP;
            async_ld16(ldsA + boff + lo0, gA0 + knext);
            async_ld16(ldsA + boff + lo1, gA1 + knext);
            async_ld16(ldsB + boff + lo0, gB0 + knext);
            async_ld16(ldsB + boff + lo1, gB1 + knext);
            asm volatile("s_wait_asynccnt 0x4" ::: "memory");  // buf[cur] landed
        } else {
            asm volatile("s_wait_asynccnt 0x0" ::: "memory");  // drain
        }
        __syncthreads();   // buf[cur] complete for the whole workgroup

        const _Float16* Ab = &As[cur][0];
        const _Float16* Bb = &Bs[cur][0];

        v16h aF;
        ((uint4*)&aF)[0] = *(const uint4*)(Ab + ar * LDS_STRIDE + abase);
        ((uint4*)&aF)[1] = *(const uint4*)(Ab + ar * LDS_STRIDE + abase + 16);

        #pragma unroll
        for (int n = 0; n < 8; ++n) {
            const int br = 16 * n + (lane & 15);
            v16h bF;
            ((uint4*)&bF)[0] = *(const uint4*)(Bb + br * LDS_STRIDE + bbase);
            ((uint4*)&bF)[1] = *(const uint4*)(Bb + br * LDS_STRIDE + bbase + 16);
            acc[n] = __builtin_amdgcn_wmma_f32_16x16x32_f16(
                false, aF, false, bF, (short)0, acc[n], false, false);
        }
    }

    // Epilogue: s = cos/T; e = exp(s - M) with M = 1/T (fixed stable shift;
    // s <= M always so no overflow). Diagonal excluded from the sums.
    const float M = TEMP_INV;
    float rsum[8];
    #pragma unroll
    for (int v = 0; v < 8; ++v) rsum[v] = 0.f;
    const int rbase = tileM + 16 * w + 8 * (lane >> 4);   // C layout: VGPR v -> M=v / 8+v

    #pragma unroll
    for (int n = 0; n < 8; ++n) {
        const int gc = tileN + 16 * n + (lane & 15);
        float csum = 0.f;
        #pragma unroll
        for (int v = 0; v < 8; ++v) {
            const float s = acc[n][v] * TEMP_INV;
            const int gr = rbase + v;
            const float e = (gr == gc) ? 0.f : __expf(s - M);
            rsum[v] += e;
            csum    += e;
        }
        csum += __shfl_xor(csum, 16);                 // fold lane-halves (same column set)
        if (lane < 16) atomicAdd(&colS[16 * n + lane], csum);   // ds_add_f32
    }

    #pragma unroll
    for (int v = 0; v < 8; ++v) {
        float r = rsum[v];
        r += __shfl_xor(r, 1);
        r += __shfl_xor(r, 2);
        r += __shfl_xor(r, 4);
        r += __shfl_xor(r, 8);                        // reduce within 16-lane half
        if ((lane & 15) == 0) atomicAdd(&rowAcc[rbase + v], r);
    }

    __syncthreads();
    if (tid < TILE) atomicAdd(&colAcc[tileN + tid], colS[tid]);
}

// ---------------------------------------------------------------------------
// Kernel 3: final reduction.
// loss = sum_i [ (log(rowAcc_i)+M - s_ii) + (log(colAcc_i)+M - s_ii) ] / (2N)
// ---------------------------------------------------------------------------
__global__ __launch_bounds__(256) void finish_kernel(
    const float* __restrict__ rowAcc, const float* __restrict__ colAcc,
    const float* __restrict__ diag, float* __restrict__ out)
{
    const int t = threadIdx.x;
    const float M = TEMP_INV;
    float local = 0.f;
    for (int i = t; i < N_ROWS; i += 256)
        local += logf(rowAcc[i]) + logf(colAcc[i]) + 2.f * M - 2.f * diag[i];

    #pragma unroll
    for (int m = 1; m < 32; m <<= 1) local += __shfl_xor(local, m);
    __shared__ float red[8];
    const int w = t >> 5, lane = t & 31;
    if (lane == 0) red[w] = local;
    __syncthreads();
    if (t == 0) {
        float total = 0.f;
        #pragma unroll
        for (int j = 0; j < 8; ++j) total += red[j];
        out[0] = total / (2.0f * (float)N_ROWS);
    }
}

// ---------------------------------------------------------------------------
extern "C" void kernel_launch(void* const* d_in, const int* in_sizes, int n_in,
                              void* d_out, int out_size, void* d_ws, size_t ws_size,
                              hipStream_t stream) {
    (void)n_in; (void)out_size; (void)ws_size; (void)in_sizes;
    const float* ecg  = (const float*)d_in[0];
    const float* text = (const float*)d_in[1];
    float* out = (float*)d_out;

    // Workspace layout
    _Float16* eN = (_Float16*)d_ws;                       // N*D halves (4 MB)
    _Float16* tN = eN + (size_t)N_ROWS * DIM;             // N*D halves (4 MB)
    float* diag   = (float*)(tN + (size_t)N_ROWS * DIM);  // N floats
    float* rowAcc = diag + N_ROWS;                        // N floats
    float* colAcc = rowAcc + N_ROWS;                      // N floats

    hipMemsetAsync(rowAcc, 0, 2 * N_ROWS * sizeof(float), stream);

    norm_diag_kernel<<<N_ROWS, 256, 0, stream>>>(ecg, text, eN, tN, diag);

    dim3 grid(N_ROWS / TILE, N_ROWS / TILE);
    ntxent_gemm_kernel<<<grid, 256, 0, stream>>>(eN, tN, rowAcc, colAcc);

    finish_kernel<<<1, 256, 0, stream>>>(rowAcc, colAcc, diag, out);
}
